// GNN_22668837388505
// MI455X (gfx1250) — matrix operand order repaired and checked by reading, not compiled, not run on hardware
//
#include <hip/hip_runtime.h>

#define NN 50000
#define NE 800000
#define MIN_F 16
#define H 160
#define BN_EPS 1e-5f
#define LDSW 168   // LDS row stride in bf16: 336B rows (16B aligned, reduced bank conflicts)

typedef __attribute__((ext_vector_type(16))) __bf16 v16bf;
typedef __attribute__((ext_vector_type(8)))  __bf16 v8bf;
typedef __attribute__((ext_vector_type(8)))  float  v8f;

static __host__ int cdiv(int a, int b) { return (a + b - 1) / b; }

__global__ void k_zero(float* __restrict__ p, int n) {
    int i = blockIdx.x * blockDim.x + threadIdx.x;
    if (i < n) p[i] = 0.0f;
}

// One thread per (edge, 4-feature chunk): gather x[src] row chunk, atomic-add into agg[dst].
// Separate row strides so layer-0 agg can be zero-padded to K=32 for the WMMA K-loop.
__global__ void k_scatter(const float* __restrict__ x, const int* __restrict__ src,
                          const int* __restrict__ dst, float* __restrict__ agg,
                          int nE, int F, int xs, int as) {
    int chunks = F >> 2;
    int i = blockIdx.x * blockDim.x + threadIdx.x;
    if (i >= nE * chunks) return;
    int e = i / chunks;
    int c = (i - e * chunks) << 2;
    int s = src[e], d = dst[e];
    float4 v = *(const float4*)(x + (size_t)s * xs + c);
    float* o = agg + (size_t)d * as + c;
    unsafeAtomicAdd(o + 0, v.x);
    unsafeAtomicAdd(o + 1, v.y);
    unsafeAtomicAdd(o + 2, v.z);
    unsafeAtomicAdd(o + 3, v.w);
}

// fp32 weights [K x H] -> bf16 [Kpad x H], zero pad rows (tiny).
__global__ void k_cvt_w(const float* __restrict__ in, __bf16* __restrict__ out,
                        int K, int Kpad, int Hd) {
    int i = blockIdx.x * blockDim.x + threadIdx.x;
    if (i >= Kpad * Hd) return;
    int k = i / Hd, n = i - k * Hd;
    out[i] = (__bf16)((k < K) ? in[(size_t)k * Hd + n] : 0.0f);
}

// C[rows x 160] = A[rows x Kpad](fp32, cvt->bf16 in-register) * W[Kpad x 160](bf16) + bias.
// One wave per 16-row strip; 10 accumulators cover all 160 columns -> 10 WMMAs per K-step.
// W staged once per block into LDS. Optional fused epilogues: BN-stat atomics OR ReLU.
__global__ void __launch_bounds__(256)
k_gemm_strip(const float* __restrict__ A, const __bf16* __restrict__ W,
             const float* __restrict__ bias, float* __restrict__ C,
             float* __restrict__ sums, float* __restrict__ sqs,
             int rows, int Kpad, int doStats, int doRelu) {
    __shared__ __bf16 lw[H * LDSW];                 // 53,760 B of the 320 KB WGP LDS

    // Cooperative W -> LDS stage: 16B chunks, both sides 16B aligned.
    int nch = (Kpad * H) >> 3;                      // chunks of 8 bf16; H/8 == 20
    for (int i = threadIdx.x; i < nch; i += 256) {
        int k = i / 20;
        int off = (i - k * 20) << 3;
        *(v8bf*)(lw + k * LDSW + off) = *(const v8bf*)(W + (size_t)k * H + off);
    }
    __syncthreads();

    int wid  = (blockIdx.x << 3) + (threadIdx.x >> 5);
    int lane = threadIdx.x & 31;
    if (wid >= (rows >> 4)) return;
    int half = lane >> 4;                           // C rows 0-7 vs 8-15; A K-subchunk select
    int l15  = lane & 15;

    v8f acc[10];
#pragma unroll
    for (int t = 0; t < 10; ++t) {
        float bv = bias[t * 16 + l15];
#pragma unroll
        for (int r = 0; r < 8; ++r) acc[t][r] = bv;
    }

    const float* arow = A + (size_t)(wid * 16 + l15) * Kpad;
    for (int k0 = 0; k0 < Kpad; k0 += 32) {
        // A fragment: lane<16 holds K {k0..+7, k0+16..+23}; lane>=16 holds {+8, +24}
        v8f f0 = *(const v8f*)(arow + k0 + half * 8);
        v8f f1 = *(const v8f*)(arow + k0 + 16 + half * 8);
        v8bf a0, a1;
#pragma unroll
        for (int r = 0; r < 8; ++r) { a0[r] = (__bf16)f0[r]; a1[r] = (__bf16)f1[r]; }
        v16bf av = __builtin_shufflevector(a0, a1, 0, 1, 2, 3, 4, 5, 6, 7,
                                           8, 9, 10, 11, 12, 13, 14, 15);
        const __bf16* brow = lw + (k0 + half * 8) * LDSW + l15;   // lane l = W column l15, K rows like A
#pragma unroll
        for (int t = 0; t < 10; ++t) {
            v16bf bm;  // K {half*8..+7, 16+half*8..+7} of column t*16+l15
#pragma unroll
            for (int s = 0; s < 16; ++s) bm[s] = brow[(s < 8 ? s : s + 8) * LDSW + t * 16];
            acc[t] = __builtin_amdgcn_wmma_f32_16x16x32_bf16(false, av, false, bm,
                                                             (short)0, acc[t], false, false);
        }
    }

    float* cbase = C + (size_t)(wid * 16 + half * 8) * H + l15;
#pragma unroll
    for (int t = 0; t < 10; ++t) {
        int col = t * 16 + l15;
        float s = 0.0f, q = 0.0f;
#pragma unroll
        for (int r = 0; r < 8; ++r) {
            float v = acc[t][r];
            if (doRelu) v = v > 0.0f ? v : 0.0f;
            cbase[(size_t)r * H + t * 16] = v;
            s += v; q += v * v;
        }
        if (doStats) {
            // lanes l and l^16 share a column: merge, then one atomic pair per column/wave
            s += __shfl_xor(s, 16, 32);
            q += __shfl_xor(q, 16, 32);
            if (half == 0) {
                unsafeAtomicAdd(&sums[col], s);
                unsafeAtomicAdd(&sqs[col], q);
            }
        }
    }
}

__global__ void k_bn_relu(float* __restrict__ h, const float* __restrict__ sums,
                          const float* __restrict__ sqs, const float* __restrict__ g,
                          const float* __restrict__ be, int rows, int Hd) {
    int i = blockIdx.x * blockDim.x + threadIdx.x;
    if (i >= rows * Hd) return;
    int c = i % Hd;
    float inv = 1.0f / (float)rows;
    float mu  = sums[c] * inv;
    float var = sqs[c] * inv - mu * mu;
    float v = (h[i] - mu) * rsqrtf(var + BN_EPS) * g[c] + be[c];
    h[i] = v > 0.0f ? v : 0.0f;
}

// One wave per row: 160 = 5 elements per lane; max/sum via lane shuffles.
__global__ void k_logsoftmax(const float* __restrict__ in, float* __restrict__ out,
                             int rows, int Hd) {
    int wid  = (blockIdx.x * blockDim.x + threadIdx.x) >> 5;
    int lane = threadIdx.x & 31;
    if (wid >= rows) return;
    const float* r = in + (size_t)wid * Hd;
    float vals[5];
    float mx = -3.0e38f;
#pragma unroll
    for (int j = 0; j < 5; ++j) { vals[j] = r[lane + j * 32]; mx = fmaxf(mx, vals[j]); }
#pragma unroll
    for (int off = 16; off; off >>= 1) mx = fmaxf(mx, __shfl_xor(mx, off, 32));
    float s = 0.0f;
#pragma unroll
    for (int j = 0; j < 5; ++j) s += __expf(vals[j] - mx);
#pragma unroll
    for (int off = 16; off; off >>= 1) s += __shfl_xor(s, off, 32);
    float lg = __logf(s);
    float* o = out + (size_t)wid * Hd;
#pragma unroll
    for (int j = 0; j < 5; ++j) o[lane + j * 32] = vals[j] - mx - lg;
}

extern "C" void kernel_launch(void* const* d_in, const int* in_sizes, int n_in,
                              void* d_out, int out_size, void* d_ws, size_t ws_size,
                              hipStream_t stream) {
    const float* x    = (const float*)d_in[0];
    const int*   esrc = (const int*)d_in[1];
    const int*   edst = (const int*)d_in[2];
    // params per layer: W1,b1,g1,be1,W2,b2,g2,be2,W3,b3
    const float* P[3][10];
    for (int l = 0; l < 3; ++l)
        for (int j = 0; j < 10; ++j)
            P[l][j] = (const float*)d_in[3 + l * 10 + j];

    size_t NH = (size_t)NN * H;                // 8,000,000
    float*  agg = (float*)d_ws;                // N x 160 (layer0 uses stride-32 prefix)
    float*  t1  = agg + NH;
    float*  t2  = t1 + NH;
    __bf16* wbf = (__bf16*)(t2 + NH);          // up to 160x160 bf16 weights
    float*  sums = (float*)(wbf + H * H);
    float*  sqs  = sums + H;

    int gemmBlocks = cdiv(NN / 16, 8);         // 1 wave per 16-row strip, 8 waves/block
    int nh = (int)NH;

    const float* xcur = x;
    for (int l = 0; l < 3; ++l) {
        int Fin  = (l == 0) ? MIN_F : H;
        int Kpad = (l == 0) ? 32 : H;
        // ---- aggregation into zero-padded agg[N x Kpad] ----
        int nAgg = NN * Kpad;
        k_zero<<<cdiv(nAgg, 256), 256, 0, stream>>>(agg, nAgg);
        int nsc = NE * (Fin >> 2);
        k_scatter<<<cdiv(nsc, 256), 256, 0, stream>>>(xcur, esrc, edst, agg,
                                                      NE, Fin, Fin, Kpad);

        // ---- Linear1 (+fused BN stats) -> BN+ReLU ----
        k_cvt_w<<<cdiv(Kpad * H, 256), 256, 0, stream>>>(P[l][0], wbf, Fin, Kpad, H);
        k_zero<<<2, 256, 0, stream>>>(sums, 2 * H);
        k_gemm_strip<<<gemmBlocks, 256, 0, stream>>>(agg, wbf, P[l][1], t1,
                                                     sums, sqs, NN, Kpad, 1, 0);
        k_bn_relu<<<cdiv(nh, 256), 256, 0, stream>>>(t1, sums, sqs, P[l][2], P[l][3], NN, H);

        // ---- Linear2 (+fused BN stats) -> BN+ReLU ----
        k_cvt_w<<<cdiv(H * H, 256), 256, 0, stream>>>(P[l][4], wbf, H, H, H);
        k_zero<<<2, 256, 0, stream>>>(sums, 2 * H);
        k_gemm_strip<<<gemmBlocks, 256, 0, stream>>>(t1, wbf, P[l][5], t2,
                                                     sums, sqs, NN, H, 1, 0);
        k_bn_relu<<<cdiv(nh, 256), 256, 0, stream>>>(t2, sums, sqs, P[l][6], P[l][7], NN, H);

        // ---- Linear3 (plain; fused ReLU for layers 0,1) ----
        k_cvt_w<<<cdiv(H * H, 256), 256, 0, stream>>>(P[l][8], wbf, H, H, H);
        k_gemm_strip<<<gemmBlocks, 256, 0, stream>>>(t2, wbf, P[l][9], t1,
                                                     sums, sqs, NN, H, 0, (l < 2) ? 1 : 0);
        if (l < 2) xcur = t1;
    }
    k_logsoftmax<<<cdiv(NN * 32, 256), 256, 0, stream>>>(t1, (float*)d_out, NN, H);
}
